// SeMCA_19447611916372
// MI455X (gfx1250) — matrix operand order repaired and checked by reading, not compiled, not run on hardware
//
#include <hip/hip_runtime.h>
#include <hip/hip_bf16.h>

typedef __attribute__((ext_vector_type(16))) _Float16 v16h;
typedef __attribute__((ext_vector_type(8)))  float    v8f;

#define BS_   128
#define C_    256
#define NPIX  20736        // 256*81
#define NG    25           // unfold positions / groups

// Fully fused SeMCA: one workgroup per batch image, all tensors resident in the
// 320KB CDNA5 LDS; attention-embed matmuls on v_wmma_f32_16x16x32_f16.
// Branchless gathers; minimal cross-lane traffic (2 ds_bpermute per tile).
__global__ __launch_bounds__(256) void semca_fused_kernel(
    const float* __restrict__ x,   const float* __restrict__ wk,  const float* __restrict__ bk,
    const float* __restrict__ w1,  const float* __restrict__ bng, const float* __restrict__ bnb,
    const float* __restrict__ bnm, const float* __restrict__ bnv,
    const float* __restrict__ w2,  const float* __restrict__ b2,
    const float* __restrict__ wvw, const float* __restrict__ bvw,
    float* __restrict__ out)
{
    __shared__ float  qk[2 * NPIX + 128];  // x[b] | k1[b] | pad for masked gathers
    __shared__ float  vls[NPIX];           // depthwise 3x3 value map
    __shared__ v16h   w1pk[NG * 32];       // W1 prepacked in A-fragment lane order
    __shared__ v16h   w2pk[NG * 32];       // W2 prepacked in A-fragment lane order
    __shared__ float4 scb4[NG * 5];        // folded BN (sc,bs) pairs, 80B/group
    __shared__ float4 b2p4[NG * 3];        // b2, 48B/group
    __shared__ float  wks[729], bks[81], wvs[2304], bvs[256];

    const int tid = threadIdx.x;
    const int b   = blockIdx.x;

    // ---- stage 0: loads + weight prepack ----
    {   const float4* xb4 = (const float4*)(x + (size_t)b * NPIX);
        float4* q4 = (float4*)qk;
        for (int i = tid; i < NPIX / 4; i += 256) q4[i] = xb4[i];
    }
    // prepack W1/W2 as 16x32 f16 A fragments (contiguous k order, same as B gather): lane L row m=L&15,
// element e -> K = (L>>4)*16 + e
    for (int t = tid; t < NG * 512; t += 256) {
        int g = t >> 9, rem = t & 511;
        int ln = rem >> 4, e = rem & 15;
        int lh = ln >> 4,  o = ln & 15;
        int j = e >> 1;
        int K = (lh << 4) + e;
        _Float16 v1h = (_Float16)0.f, v2h = (_Float16)0.f;
        if (o < 9) {
            if (K < 18) v1h = (_Float16)w1[g * 162 + o * 18 + K];
            if (K < 9)  v2h = (_Float16)w2[g * 81  + o * 9  + K];
        }
        ((_Float16*)w1pk)[t] = v1h;
        ((_Float16*)w2pk)[t] = v2h;
    }
    for (int i = tid; i < 225; i += 256) {
        int g = i / 9, o = i - g * 9;
        float s = bng[i] * rsqrtf(bnv[i] + 1e-5f);          // fold BN (eval)
        ((float*)scb4)[g * 20 + o * 2 + 0] = s;
        ((float*)scb4)[g * 20 + o * 2 + 1] = bnb[i] - bnm[i] * s;
        ((float*)b2p4)[g * 12 + o] = b2[i];
    }
    for (int i = tid; i < 729; i += 256) wks[i] = wk[i];
    for (int i = tid; i < 81;  i += 256) bks[i] = bk[i];
    {   const float4* w4 = (const float4*)wvw; float4* d4 = (float4*)wvs;
        for (int i = tid; i < 576; i += 256) d4[i] = w4[i];
    }
    for (int i = tid; i < 256; i += 256) bvs[i] = bvw[i];
    __syncthreads();

    // ---- stage 1: k1 (depthwise spectral conv) and v (depthwise 3x3), branchless ----
    for (int i = tid; i < NPIX; i += 256) {
        int c = i / 81, p = i - c * 81;
        float acc = bks[p];
        #pragma unroll
        for (int d = 0; d < 9; ++d) {
            int cc = c + d - 4;
            bool ok = (unsigned)cc < 256u;
            float t0 = qk[ok ? cc * 81 + p : p];
            acc = fmaf(ok ? t0 : 0.f, wks[p * 9 + d], acc);
        }
        qk[NPIX + i] = acc;

        int yy = p / 9, xx = p - yy * 9;
        float va = bvs[c];
        #pragma unroll
        for (int ki = 0; ki < 3; ++ki)
            #pragma unroll
            for (int kj = 0; kj < 3; ++kj) {
                int y2 = yy + ki - 1, x2 = xx + kj - 1;
                bool ok = ((unsigned)y2 < 9u) & ((unsigned)x2 < 9u);
                float t0 = qk[c * 81 + (ok ? y2 * 9 + x2 : 0)];
                va = fmaf(ok ? t0 : 0.f, wvs[c * 9 + ki * 3 + kj], va);
            }
        vls[i] = va;
    }
    __syncthreads();

    // ---- stage 2: per-group WMMA embed + softmax + combine ----
    const int lane  = tid & 31;
    const int wavid = tid >> 5;       // wave owns channel tiles 2w, 2w+1
    const int lhalf = lane >> 4;
    const int l15   = lane & 15;

    // g-invariant gather codes: pc = (base + kh5*9 + kw5)<<8 | kh5<<4 | kw5
    // (kh5=15 poisons K>=18; kh5/kw5==10 self-mask vs thresholds <=9)
    int pcd[2][16];
    int cb81[2];
    #pragma unroll
    for (int ts = 0; ts < 2; ++ts) {
        int cch = (wavid * 2 + ts) * 16 + l15;
        cb81[ts] = cch * 81;
        #pragma unroll
        for (int e = 0; e < 16; ++e) {
            int K  = (lhalf << 4) + e;
            int Kc = K < 18 ? K : 17;
            int which = (Kc >= 9);
            int jj = Kc - which * 9;
            int f  = jj * 256 + cch;
            int ch = f / 9, k = f - ch * 9;
            int kh = k / 3, kw = k - kh * 3;
            int kh5  = (K < 18) ? kh * 5 : 15;
            int kw5  = kw * 5;
            int adr0 = which * NPIX + ch * 81 + kh5 * 9 + kw5;
            pcd[ts][e] = (adr0 << 8) | (kh5 << 4) | kw5;
        }
    }

    for (int g = 0; g < NG; ++g) {
        const int oh = g / 5, ow = g - (g / 5) * 5;
        const int soff = oh * 9 + ow;          // scalar per-g address offset
        const int thrH = 9 - oh, thrW = 9 - ow;
        const v16h a1 = w1pk[g * 32 + lane];   // 2x ds_load_b128
        const v16h a2 = w2pk[g * 32 + lane];

        // hoisted per-g constants (registers, shared by both tiles)
        float4 s0 = scb4[g * 5 + 0], s1 = scb4[g * 5 + 1], s2 = scb4[g * 5 + 2],
               s3 = scb4[g * 5 + 3], s4 = scb4[g * 5 + 4];
        float scr[9] = {s0.x, s0.z, s1.x, s1.z, s2.x, s2.z, s3.x, s3.z, s4.x};
        float bsr[9] = {s0.y, s0.w, s1.y, s1.w, s2.y, s2.w, s3.y, s3.w, s4.y};
        float4 bb0 = b2p4[g * 3 + 0], bb1 = b2p4[g * 3 + 1], bb2 = b2p4[g * 3 + 2];
        float b2v[9] = {bb0.x, bb0.y, bb0.z, bb0.w, bb1.x, bb1.y, bb1.z, bb1.w, bb2.x};

        #pragma unroll
        for (int ts = 0; ts < 2; ++ts) {
            // B1 gather: branchless unfold of {x, k1}
            v16h b1;
            #pragma unroll
            for (int e = 0; e < 16; ++e) {
                int pc = pcd[ts][e];
                bool ok = (((pc >> 4) & 15) < thrH) & ((pc & 15) < thrW);
                float t0 = qk[(pc >> 8) + soff];          // stays inside padded LDS
                b1[e] = (_Float16)(ok ? t0 : 0.f);
            }
            v8f cz = {};
            v8f h1 = __builtin_amdgcn_wmma_f32_16x16x32_f16(
                         false, a1, false, b1, (short)0, cz, false, false);

            // folded BN + ReLU; only r==0 is consumed from the upper half
            float hb[8];
            hb[0] = fmaxf(0.f, fmaf(h1[0], lhalf ? scr[8] : scr[0],
                                            lhalf ? bsr[8] : bsr[0]));
            #pragma unroll
            for (int r = 1; r < 8; ++r)
                hb[r] = fmaxf(0.f, fmaf(h1[r], scr[r], bsr[r]));

            // D-layout -> B-layout: rows 0-7 are already local; row 8 via 1 shuffle
            float h8 = __shfl(hb[0], l15 + 16, 32);
            v16h b2f;
            #pragma unroll
            for (int e = 0; e < 16; ++e) b2f[e] = (_Float16)0.f;
            #pragma unroll
            for (int e = 0; e < 8; ++e) b2f[e] = (_Float16)(lhalf ? 0.f : hb[e]);
            b2f[8] = (_Float16)(lhalf ? 0.f : h8);

            v8f d2 = __builtin_amdgcn_wmma_f32_16x16x32_f16(
                         false, a2, false, b2f, (short)0, cz, false, false);

            // logits: rows 0-7 local, row 8 via 1 shuffle; then softmax
            float d8 = __shfl(d2[0], l15 + 16, 32);
            float lv[9];
            #pragma unroll
            for (int dd = 0; dd < 8; ++dd) lv[dd] = d2[dd] + b2v[dd];
            lv[8] = d8 + b2v[8];
            float mx = lv[0];
            #pragma unroll
            for (int dd = 1; dd < 9; ++dd) mx = fmaxf(mx, lv[dd]);
            float ssum = 0.f;
            #pragma unroll
            for (int dd = 0; dd < 9; ++dd) { lv[dd] = __expf(lv[dd] - mx); ssum += lv[dd]; }
            float inv = 1.f / ssum;
            #pragma unroll
            for (int dd = 0; dd < 9; ++dd) lv[dd] *= inv;

            // combine: this group's output pixels ((row%5,col%5)==(oh,ow), row,col in [3,12))
            if (lhalf == 0) {
                const int cch = (wavid * 2 + ts) * 16 + l15;
                const int cb  = cb81[ts];
                #pragma unroll
                for (int ri = 0; ri < 3; ++ri) {
                    int row = oh + 5 * ri;
                    if (row < 3 || row >= 12) continue;      // uniform branch
                    #pragma unroll
                    for (int ci = 0; ci < 3; ++ci) {
                        int col = ow + 5 * ci;
                        if (col < 3 || col >= 12) continue;  // uniform branch
                        float acc = 0.f;
                        if (row >= 6 && col >= 6) {          // uniform: v_unf pad top/left 6
                            int pp = (row - 6) * 9 + (col - 6);
                            #pragma unroll
                            for (int dd = 0; dd < 9; ++dd) {
                                int chd = cch + dd - 4;      // spectral window +-4
                                bool ok = (unsigned)chd < 256u;
                                float t0 = vls[ok ? chd * 81 + pp : pp];
                                acc = fmaf(ok ? t0 : 0.f, lv[dd], acc);
                            }
                        }
                        if (row < 9 && col < 9)              // k1_pad residual
                            acc += qk[NPIX + cb + row * 9 + col];
                        out[((size_t)b * C_ + cch) * 81 + (row - 3) * 9 + (col - 3)] = acc;
                    }
                }
            }
        } // ts
    } // g
}

extern "C" void kernel_launch(void* const* d_in, const int* in_sizes, int n_in,
                              void* d_out, int out_size, void* d_ws, size_t ws_size,
                              hipStream_t stream) {
    (void)in_sizes; (void)n_in; (void)d_ws; (void)ws_size; (void)out_size;
    const float* x   = (const float*)d_in[0];
    const float* wk  = (const float*)d_in[1];
    const float* bk  = (const float*)d_in[2];
    const float* w1  = (const float*)d_in[3];
    const float* bng = (const float*)d_in[4];
    const float* bnb = (const float*)d_in[5];
    const float* bnm = (const float*)d_in[6];
    const float* bnv = (const float*)d_in[7];
    const float* w2  = (const float*)d_in[8];
    const float* b2  = (const float*)d_in[9];
    const float* wv  = (const float*)d_in[10];
    const float* bv  = (const float*)d_in[11];
    float* out = (float*)d_out;

    semca_fused_kernel<<<dim3(BS_), dim3(256), 0, stream>>>(
        x, wk, bk, w1, bng, bnb, bnm, bnv, w2, b2, wv, bv, out);
}
